// TimeAlignedPredictionNetwork_58411555225675
// MI455X (gfx1250) — matrix operand order repaired and checked by reading, not compiled, not run on hardware
//
#include <hip/hip_runtime.h>

// ---------------------------------------------------------------------------
// Types for CDNA5 WMMA (wave32): v_wmma_f32_16x16x32_bf16
//   A: 16x32 bf16  -> 16 bf16 / lane (8 VGPRs)
//   B: 32x16 bf16  -> 16 bf16 / lane (8 VGPRs)
//   C/D: 16x16 f32 ->  8 f32  / lane (8 VGPRs)
// ---------------------------------------------------------------------------
typedef __bf16 bf16;
typedef __attribute__((ext_vector_type(16))) __bf16 v16bf;
typedef __attribute__((ext_vector_type(8)))  float  v8f;

union Frag16 {       // 16 bf16 = 32 bytes, loaded as two 16B chunks
    v16bf v;
    uint4 q[2];
};

#define NP     12
#define BB     8
#define SC     512
#define SP     524
#define NCAND  4
#define DAR    512
#define DENC   512
#define DVAL   510          // dimEnc - 2 used in the final dot
#define SBLK   64           // s-rows per workgroup
#define CT_STR 520          // LDS row stride in bf16 (512 + 8 pad)

// ---------------------------------------------------------------------------
// Prep 1: pl = sigmoid(predictedLengths), pls = cumsum(pl)   [B, SP]
// ---------------------------------------------------------------------------
__global__ void prep_sigmoid_cumsum(const float* __restrict__ pred,
                                    float* __restrict__ pl,
                                    float* __restrict__ pls) {
    int b = blockIdx.x;
    if (threadIdx.x == 0) {
        float run = 0.0f;
        for (int j = 0; j < SP; ++j) {
            float x = pred[b * SP + j];
            float s = 1.0f / (1.0f + expf(-x));
            pl[b * SP + j] = s;
            run += s;
            pls[b * SP + j] = run;
        }
    }
}

// ---------------------------------------------------------------------------
// Prep 2: split W (f32) into bf16 hi/lo pair (3-term bf16 GEMM ~ fp32 accuracy).
// Rows e >= 510 are zeroed: their preds are never used by the reference
// (it slices [..., :510]), and zeroing them lets the main kernel issue
// unconditional (clamped) candidate loads with no exec-mask branching.
// ---------------------------------------------------------------------------
__global__ void prep_w_split(const float* __restrict__ W,
                             bf16* __restrict__ Whi,
                             bf16* __restrict__ Wlo, int n) {
    int i = blockIdx.x * blockDim.x + threadIdx.x;
    if (i < n) {
        int e = (i >> 9) & (DENC - 1);   // i = ((k*DENC)+e)*DAR + a
        float w = (e < DVAL) ? W[i] : 0.0f;
        bf16 h = (bf16)w;
        Whi[i] = h;
        Wlo[i] = (bf16)(w - (float)h);
    }
}

// ---------------------------------------------------------------------------
// Main fused kernel: per workgroup = one (k, b, 64-wide s block).
//   preds[k] = c2 @ W[k]^T via WMMA (hi/lo split), fused with the
//   candidates dot so preds never hits memory.
// Grid: 12*8*8 = 768 blocks, 256 threads (8 wave32).
// ---------------------------------------------------------------------------
__global__ __launch_bounds__(256, 4) void tapn_main(
    const float* __restrict__ c,        // [B, SC, DAR]
    const float* __restrict__ cand,     // [NP, B, NCAND, SC, DVAL]
    const bf16*  __restrict__ Whi,      // [NP, DENC, DAR]
    const bf16*  __restrict__ Wlo,      // [NP, DENC, DAR]
    const float* __restrict__ pl,       // [B, SP]
    const float* __restrict__ pls,      // [B, SP]
    float*       __restrict__ out)      // [NP, B, NCAND, SC]
{
    __shared__ bf16 ctH[SBLK * CT_STR];   // c2^T tile, hi part  (row = s, col = a)
    __shared__ bf16 ctL[SBLK * CT_STR];   // c2^T tile, lo part
    __shared__ float oacc[NCAND * SBLK];  // output accumulator [n][s_local]

    const int bid = blockIdx.x;
    const int k  = bid >> 6;            // / (8*8)
    const int b  = (bid >> 3) & 7;
    const int s0 = (bid & 7) * SBLK;

    const int t = threadIdx.x;
    oacc[t] = 0.0f;                      // 4*64 == 256 entries

    // ---- Stage c2 tile into LDS (patch channels 510=pls, 511=pl; bf16 split)
    const float* cbase = c + ((size_t)b * SC + s0) * DAR;
    for (int it = 0; it < (SBLK * DAR) / 1024; ++it) {   // 32 iters
        int f  = it * 1024 + t * 4;
        int sl = f >> 9;                 // / 512
        int a  = f & 511;
        float4 v = *(const float4*)(cbase + (size_t)sl * DAR + a);
        if (a == 508) {
            v.z = pls[b * SP + s0 + sl];  // channel 510
            v.w = pl [b * SP + s0 + sl];  // channel 511
        }
        float vv[4] = {v.x, v.y, v.z, v.w};
        #pragma unroll
        for (int j = 0; j < 4; ++j) {
            bf16 h = (bf16)vv[j];
            ctH[sl * CT_STR + a + j] = h;
            ctL[sl * CT_STR + a + j] = (bf16)(vv[j] - (float)h);
        }
    }
    __syncthreads();

    const int wid  = t >> 5;             // wave 0..7 -> e-range [wid*64, wid*64+64)
    const int lane = t & 31;
    const int m    = lane & 15;
    const int h    = lane >> 4;

    // Per-(k,b) candidates base; row offset added per (n, ss).
    const float* candkb = cand + (((size_t)k * BB + b) * NCAND) * SC * DVAL;

    float part[NCAND][4];                // [n][ssub] dot partials
    #pragma unroll
    for (int n = 0; n < NCAND; ++n)
        #pragma unroll
        for (int ss = 0; ss < 4; ++ss) part[n][ss] = 0.0f;

    #pragma unroll 1
    for (int et = 0; et < 4; ++et) {
        const int e0 = wid * 64 + et * 16;

        v8f acc[4];
        #pragma unroll
        for (int ss = 0; ss < 4; ++ss) acc[ss] = (v8f)0.0f;

        // A-fragment row for this lane (W[k] row e0+m, contiguous in a)
        const bf16* whr = Whi + ((size_t)k * DENC + e0 + m) * DAR;
        const bf16* wlr = Wlo + ((size_t)k * DENC + e0 + m) * DAR;

        // ---- GEMM over K = a = 0..511 in steps of 32.
        // unroll 1: keep ds_load B-fragments in-loop (no hoist-then-spill).
        #pragma unroll 1
        for (int a0 = 0; a0 < DAR; a0 += 32) {
            Frag16 ah, al;
            ah.q[0] = *(const uint4*)(whr + a0 + 8 * h);
            ah.q[1] = *(const uint4*)(whr + a0 + 16 + 8 * h);
            al.q[0] = *(const uint4*)(wlr + a0 + 8 * h);
            al.q[1] = *(const uint4*)(wlr + a0 + 16 + 8 * h);

            #pragma unroll
            for (int ss = 0; ss < 4; ++ss) {
                const int srow = ss * 16 + m;
                Frag16 bh, bl;
                bh.q[0] = *(const uint4*)&ctH[srow * CT_STR + a0 + 8 * h];
                bh.q[1] = *(const uint4*)&ctH[srow * CT_STR + a0 + 16 + 8 * h];
                bl.q[0] = *(const uint4*)&ctL[srow * CT_STR + a0 + 8 * h];
                bl.q[1] = *(const uint4*)&ctL[srow * CT_STR + a0 + 16 + 8 * h];

                // 3-term split: hi*hi + hi*lo + lo*hi  (~fp32 accuracy)
                acc[ss] = __builtin_amdgcn_wmma_f32_16x16x32_bf16(
                    false, ah.v, false, bh.v, (short)0, acc[ss], false, false);
                acc[ss] = __builtin_amdgcn_wmma_f32_16x16x32_bf16(
                    false, ah.v, false, bl.v, (short)0, acc[ss], false, false);
                acc[ss] = __builtin_amdgcn_wmma_f32_16x16x32_bf16(
                    false, al.v, false, bh.v, (short)0, acc[ss], false, false);
            }
        }

        // ---- Fused candidates dot for this e-tile.
        // D layout: lane, vgpr i  <->  preds[e0 + i + 8h][s0 + 16*ssub + m]
        // Clamped offsets (no predication): the single chunk that would read
        // e=510 re-reads e=508 instead; its acc lanes are zero (W rows >=510
        // zeroed in prep), so it contributes nothing.
        int ecl[4];
        #pragma unroll
        for (int j = 0; j < 4; ++j) {
            int e = e0 + 8 * h + 2 * j;
            ecl[j] = (e < DVAL) ? e : (DVAL - 2);
        }

        #pragma unroll
        for (int n = 0; n < NCAND; ++n) {
            #pragma unroll
            for (int ss = 0; ss < 4; ++ss) {
                const int sg = s0 + ss * 16 + m;
                const float* crow = candkb + ((size_t)(n * SC + sg)) * DVAL;
                float p = 0.0f;
                #pragma unroll
                for (int j = 0; j < 4; ++j) {
                    float2 cv = *(const float2*)(crow + ecl[j]);
                    p += acc[ss][2 * j]     * cv.x;
                    p += acc[ss][2 * j + 1] * cv.y;
                }
                part[n][ss] += p;
            }
        }
    }

    // ---- Combine across waves / lane-halves via LDS float atomics
    #pragma unroll
    for (int n = 0; n < NCAND; ++n)
        #pragma unroll
        for (int ss = 0; ss < 4; ++ss)
            atomicAdd(&oacc[n * SBLK + ss * 16 + m], part[n][ss]);

    __syncthreads();

    // ---- Write out[k, b, n, s] = oacc / 510
    {
        const int n  = t >> 6;
        const int sl = t & 63;
        out[((((size_t)k * BB + b) * NCAND + n) * SC) + s0 + sl] =
            oacc[t] * (1.0f / (float)DVAL);
    }
}

// ---------------------------------------------------------------------------
// Harness entry
// ---------------------------------------------------------------------------
extern "C" void kernel_launch(void* const* d_in, const int* in_sizes, int n_in,
                              void* d_out, int out_size, void* d_ws, size_t ws_size,
                              hipStream_t stream) {
    const float* c    = (const float*)d_in[0];   // [8,512,512]
    const float* cand = (const float*)d_in[1];   // [12,8,4,512,510]
    const float* pred = (const float*)d_in[2];   // [8,524]
    const float* W    = (const float*)d_in[3];   // [12,512,512]
    float* out = (float*)d_out;

    char*  ws  = (char*)d_ws;
    float* pl  = (float*)ws;                      // 8*524 f32
    float* pls = pl + BB * SP;                    // 8*524 f32
    bf16*  Whi = (bf16*)(ws + 40960);             // 12*512*512 bf16
    bf16*  Wlo = Whi + (size_t)NP * DENC * DAR;   // 12*512*512 bf16

    prep_sigmoid_cumsum<<<BB, 32, 0, stream>>>(pred, pl, pls);

    const int nW = NP * DENC * DAR;
    prep_w_split<<<(nW + 255) / 256, 256, 0, stream>>>(W, Whi, Wlo, nW);

    tapn_main<<<NP * BB * (SC / SBLK), 256, 0, stream>>>(
        c, cand, Whi, Wlo, pl, pls, out);
}
